// MLPPredictor_54116587929903
// MI455X (gfx1250) — matrix operand order repaired and checked by reading, not compiled, not run on hardware
//
#include <hip/hip_runtime.h>

typedef __attribute__((ext_vector_type(16))) _Float16 v16h;
typedef __attribute__((ext_vector_type(8)))  float    v8f;
typedef __attribute__((ext_vector_type(4)))  float    v4f;

#define HID       128
#define TWO_H     256
#define EPW       16      // edges per wave (M tile)
#define WPB       8       // waves per block
#define EPB       (EPW*WPB)

__global__ __launch_bounds__(256, 2)
void edge_mlp_wmma(const int* __restrict__ src,
                   const int* __restrict__ dst,
                   const float* __restrict__ feats,
                   const float* __restrict__ W1,
                   const float* __restrict__ b1,
                   const float* __restrict__ W2,
                   const float* __restrict__ b2,
                   float* __restrict__ out,
                   int E)
{
    // W1 staged as pre-swizzled f16 WMMA B-fragments:
    // B[k][n] = W1[n][k]; for (kt,nt), lane l: halfs e=0..15 hold
    // K = kt*32 + (l>=16?16:0) + e, col = nt*16 + (l&15).
    __shared__ __align__(32) _Float16 Blds[8][8][32][16];   // 64 KB

    const int tid = threadIdx.x;

    // ---- stage W1 (fp32 -> f16, WMMA B layout) ----
    #pragma unroll
    for (int it = 0; it < 8; ++it) {
        int t   = tid + it * 256;        // 2048 (kt,nt,lane) triples
        int kt  = t >> 8;
        int nt  = (t >> 5) & 7;
        int ln  = t & 31;
        int kof = (ln >= 16) ? 16 : 0;
        int col = nt * 16 + (ln & 15);
        const float* w = W1 + (long long)col * TWO_H + kt * 32 + kof;
        v4f r0 = *(const v4f*)(w + 0);
        v4f r1 = *(const v4f*)(w + 4);
        v4f r2 = *(const v4f*)(w + 8);
        v4f r3 = *(const v4f*)(w + 12);
        float tmp[16];
        *(v4f*)&tmp[0]  = r0;
        *(v4f*)&tmp[4]  = r1;
        *(v4f*)&tmp[8]  = r2;
        *(v4f*)&tmp[12] = r3;
        _Float16* o = &Blds[kt][nt][ln][0];
        #pragma unroll
        for (int e = 0; e < 16; ++e) o[e] = (_Float16)tmp[e];
    }
    __syncthreads();

    const int  wave = tid >> 5;
    const int  lane = tid & 31;
    const int  row  = lane & 15;           // edge within M tile
    const bool hi   = lane >= 16;

    const long long ebase = (long long)blockIdx.x * EPB + wave * EPW;
    long long eclamped = ebase + row;
    if (eclamped > E - 1) eclamped = E - 1;          // clamp tail (EXEC stays full for WMMA)

    const int sn = src[eclamped];
    const int dn = dst[eclamped];
    const float* prs = feats + (long long)sn * HID;
    const float* prd = feats + (long long)dn * HID;

    v8f acc[8];
    #pragma unroll
    for (int n = 0; n < 8; ++n)
        #pragma unroll
        for (int j = 0; j < 8; ++j) acc[n][j] = 0.0f;

    const int k8 = hi ? 16 : 0;   // A-layout: lane half offsets K by +16 (contiguous 16-run)

    // ---- layer 1: [16 x 256] x [256 x 128] via 8x8 WMMA tiles ----
    #pragma unroll
    for (int kt = 0; kt < 8; ++kt) {
        const float* p = (kt < 4) ? (prs + kt * 32) : (prd + (kt - 4) * 32);
        // A fragment 16x32 f16: e=0..15 -> K=kb+(hi?16:0)+e
        v4f x0 = *(const v4f*)(p + k8 + 0);
        v4f x1 = *(const v4f*)(p + k8 + 4);
        v4f x2 = *(const v4f*)(p + k8 + 8);
        v4f x3 = *(const v4f*)(p + k8 + 12);
        float xa[16];
        *(v4f*)&xa[0]  = x0;
        *(v4f*)&xa[4]  = x1;
        *(v4f*)&xa[8]  = x2;
        *(v4f*)&xa[12] = x3;
        v16h a;
        #pragma unroll
        for (int e = 0; e < 16; ++e) a[e] = (_Float16)xa[e];

        #pragma unroll
        for (int nt = 0; nt < 8; ++nt) {
            v16h b = *(const v16h*)&Blds[kt][nt][lane][0];
            acc[nt] = __builtin_amdgcn_wmma_f32_16x16x32_f16(
                /*neg_a=*/false, a, /*neg_b=*/false, b,
                /*c_mod=*/(short)0, acc[nt],
                /*reuse_a=*/false, /*reuse_b=*/false);
        }
    }

    // ---- layer 2 + sigmoid fused in epilogue ----
    // acc[nt][r] holds h[m][c] with m = r + (hi?8:0), c = nt*16 + (lane&15)
    const int   cb    = lane & 15;
    const float bias2 = b2[0];

    #pragma unroll
    for (int r = 0; r < 8; ++r) {
        float s = 0.0f;
        #pragma unroll
        for (int nt = 0; nt < 8; ++nt) {
            int   col = nt * 16 + cb;
            float hv  = acc[nt][r] + b1[col];
            hv = hv > 0.0f ? hv : 0.0f;      // ReLU
            s += hv * W2[col];
        }
        // reduce over the 16 lanes of each half-wave (wave32: bit4 untouched)
        s += __shfl_xor(s, 1, 32);
        s += __shfl_xor(s, 2, 32);
        s += __shfl_xor(s, 4, 32);
        s += __shfl_xor(s, 8, 32);

        float score = s + bias2;
        float sig   = 1.0f / (1.0f + __expf(-score));

        long long eo = ebase + (hi ? 8 : 0) + r;
        if (cb == r && eo < (long long)E) out[eo] = sig;
    }
}

extern "C" void kernel_launch(void* const* d_in, const int* in_sizes, int n_in,
                              void* d_out, int out_size, void* d_ws, size_t ws_size,
                              hipStream_t stream) {
    const int*   src   = (const int*)d_in[0];
    const int*   dst   = (const int*)d_in[1];
    const float* feats = (const float*)d_in[2];
    const float* W1    = (const float*)d_in[3];
    const float* b1    = (const float*)d_in[4];
    const float* W2    = (const float*)d_in[5];
    const float* b2    = (const float*)d_in[6];
    float*       out   = (float*)d_out;

    const int E = in_sizes[0];
    const int blocks = (E + EPB - 1) / EPB;
    edge_mlp_wmma<<<blocks, 256, 0, stream>>>(src, dst, feats, W1, b1, W2, b2, out, E);
}
